// CRFDecoder_37873021616561
// MI455X (gfx1250) — matrix operand order repaired and checked by reading, not compiled, not run on hardware
//
#include <hip/hip_runtime.h>
#include <stdint.h>

// CRF forward (partition function) on gfx1250.
// log_observation: [B=64, T=4096, S=128] f32, log_transition: [S,S] f32.
// out: [B] f32 = -log Z.
//
// Exp-domain rescaled scan. Per step, per 16-batch tile:
//   m[r]   = max_k fv[r,k]                      (intra-wave shfl_xor reduce)
//   p[r,k] = exp2((fv[r,k]-m[r])*log2e)         (bf16 in LDS)
//   q      = p (16x128) x E (128x128)           (4x v_wmma_f32_16x16x32_bf16,
//                                                E resident in VGPRs)
//   fv'    = m[r] + log2(q)*ln2 + obs[t]
// Grid = B/16 = 4 blocks x 256 threads (8 waves; each wave owns one N tile).

#define CRF_B 64
#define CRF_T 4096
#define CRF_S 128
#define LOG2E 1.4426950408889634f
#define LN2   0.6931471805599453f

typedef __attribute__((ext_vector_type(16))) __bf16 v16bf;
typedef __attribute__((ext_vector_type(8)))  float  v8f;

__device__ __forceinline__ uint32_t f2bf_bits(float f) {
  uint32_t u = __builtin_bit_cast(uint32_t, f);
  return (u + 0x7FFFu + ((u >> 16) & 1u)) >> 16;   // round-to-nearest-even
}
__device__ __forceinline__ uint32_t pack2bf(float a, float b) {
  return (f2bf_bits(a) & 0xFFFFu) | (f2bf_bits(b) << 16);
}

__global__ __launch_bounds__(256, 2)
void crf_forward_wmma(const float* __restrict__ obs,
                      const float* __restrict__ lt,
                      float* __restrict__ out) {
  constexpr int FSTR   = 132;   // fv row stride in floats (528B, 16B aligned)
  constexpr int PSTR32 = 68;    // p row stride in u32 (272B, 16B aligned)

  __shared__ __align__(16) float fvBuf[16 * FSTR];      // fv[16][128] f32
  __shared__ __align__(16) uint32_t pBuf[16 * PSTR32];  // p[16][128] bf16
  __shared__ float pmax[16 * 16];                       // final reduce only
  __shared__ float psum[16 * 16];                       // final reduce only
  __shared__ float rowMax[16];

  const int tid = threadIdx.x;
  const int b0  = blockIdx.x * 16;

  // reduction-view indices: 16 threads (contiguous lanes) per row, 8 cols each
  const int r  = tid >> 4;
  const int cs = tid & 15;
  const int c0 = cs * 8;

  // wmma-view indices
  const int wave = tid >> 5;
  const int lane = tid & 31;
  const int hl   = lane >> 4;     // lane half (0 / 1)
  const int ln   = lane & 15;
  const int n0   = wave * 16;     // this wave's N tile

  // ---- Preload B operands once: Eb[kb] = bf16(exp(log_transition)) ----
  // B k order used here (32x16 per op, contiguous, same as A): lanes 0-15 -> N=ln, K=2v..2v+1 (K 0..15);
  // lanes 16-31 -> same N, K 16..31.
  v16bf Eb[4];
  #pragma unroll
  for (int kb = 0; kb < 4; ++kb) {
    union { uint32_t u[8]; v16bf v; } eu;
    #pragma unroll
    for (int j = 0; j < 8; ++j) {
      const int k = kb * 32 + hl * 16 + 2 * j;
      const float e0 = __builtin_amdgcn_exp2f(lt[(size_t)k * CRF_S + n0 + ln] * LOG2E);
      const float e1 = __builtin_amdgcn_exp2f(lt[(size_t)(k + 1) * CRF_S + n0 + ln] * LOG2E);
      eu.u[j] = pack2bf(e0, e1);
    }
    Eb[kb] = eu.v;
  }

  // ---- t = 0: fv = obs[:, 0, :] ----
  {
    const float* ob = obs + (size_t)(b0 + r) * CRF_T * CRF_S + c0;
    #pragma unroll
    for (int i = 0; i < 8; ++i) fvBuf[r * FSTR + c0 + i] = ob[i];
  }
  __syncthreads();

  const size_t rowStride = (size_t)CRF_T * CRF_S;           // per-batch stride
  const size_t obsBase   = (size_t)(b0 + hl * 8) * rowStride;

  for (int t = 1; t < CRF_T; ++t) {
    // ---------- stage A: row max (shfl reduce), rescale, exp -> bf16 p ----
    float fvl[8];
    {
      const float4* fp = (const float4*)(fvBuf + r * FSTR + c0);
      float4 x0 = fp[0], x1 = fp[1];
      fvl[0] = x0.x; fvl[1] = x0.y; fvl[2] = x0.z; fvl[3] = x0.w;
      fvl[4] = x1.x; fvl[5] = x1.y; fvl[6] = x1.z; fvl[7] = x1.w;
    }
    float m = fvl[0];
    #pragma unroll
    for (int i = 1; i < 8; ++i) m = fmaxf(m, fvl[i]);
    // reduce across the 16 lanes of this row; xor masks < 16 never cross
    // the 16-lane half, so all lanes end with their row's max.
    #pragma unroll
    for (int off = 1; off < 16; off <<= 1) m = fmaxf(m, __shfl_xor(m, off));
    if (cs == 0) rowMax[r] = m;
    #pragma unroll
    for (int i = 0; i < 4; ++i) {
      const float p0 = __builtin_amdgcn_exp2f((fvl[2 * i]     - m) * LOG2E);
      const float p1 = __builtin_amdgcn_exp2f((fvl[2 * i + 1] - m) * LOG2E);
      pBuf[r * PSTR32 + (c0 >> 1) + i] = pack2bf(p0, p1);
    }
    __syncthreads();

    // ---------- stage B: q = p x E via WMMA, fv' = m + ln(q) + obs --------
    // Prefetch all 4 A operands as one DS clause, then 4 back-to-back WMMAs.
    // A k order used here (16x32 bf16, contiguous, same as Eb): lanes 0-15 hold row=ln, K in {0..15};
    // lanes 16-31 hold K in {16..31}.
    uint4 aLo[4], aHi[4];
    #pragma unroll
    for (int kb = 0; kb < 4; ++kb) {
      const uint4* pa = (const uint4*)((const char*)pBuf +
                                       ln * (PSTR32 * 4) + kb * 64 + hl * 32);
      aLo[kb] = pa[0];
      aHi[kb] = pa[1];   // +16 bytes = next 8 K
    }
    v8f acc = {};
    #pragma unroll
    for (int kb = 0; kb < 4; ++kb) {
      union { uint4 q[2]; v16bf v; } au;
      au.q[0] = aLo[kb];
      au.q[1] = aHi[kb];
      acc = __builtin_amdgcn_wmma_f32_16x16x32_bf16(
                false, au.v, false, Eb[kb], (short)0, acc, false, false);
    }

    const float* op = obs + obsBase + (size_t)t * CRF_S + (n0 + ln);
    #pragma unroll
    for (int v = 0; v < 8; ++v) {
      const int row = v + hl * 8;              // C/D layout: VGPR v, lane half
      const float base = rowMax[row] + op[(size_t)v * rowStride];
      fvBuf[row * FSTR + n0 + ln] = fmaf(__builtin_amdgcn_logf(acc[v]), LN2, base);
    }
    if (t + 8 < CRF_T) {
      __builtin_prefetch(op + 8 * CRF_S, 0, 0);   // global_prefetch_b8, t+8
    }
    __syncthreads();
  }

  // ---------- final: out[b] = -(max + ln(sum exp(fv - max))) ----------
  float m = -3.4e38f;
  #pragma unroll
  for (int i = 0; i < 8; ++i) m = fmaxf(m, fvBuf[r * FSTR + c0 + i]);
  pmax[r * 16 + cs] = m;
  __syncthreads();
  float mr = pmax[r * 16];
  #pragma unroll
  for (int i = 1; i < 16; ++i) mr = fmaxf(mr, pmax[r * 16 + i]);
  float s = 0.f;
  #pragma unroll
  for (int i = 0; i < 8; ++i)
    s += __builtin_amdgcn_exp2f((fvBuf[r * FSTR + c0 + i] - mr) * LOG2E);
  psum[r * 16 + cs] = s;
  __syncthreads();
  if (tid < 16) {
    float ss = 0.f;
    #pragma unroll
    for (int i = 0; i < 16; ++i) ss += psum[tid * 16 + i];
    float mm = pmax[tid * 16];
    #pragma unroll
    for (int i = 1; i < 16; ++i) mm = fmaxf(mm, pmax[tid * 16 + i]);
    out[b0 + tid] = -(mm + __builtin_amdgcn_logf(ss) * LN2);
  }
}

extern "C" void kernel_launch(void* const* d_in, const int* in_sizes, int n_in,
                              void* d_out, int out_size, void* d_ws, size_t ws_size,
                              hipStream_t stream) {
  (void)in_sizes; (void)n_in; (void)out_size; (void)d_ws; (void)ws_size;
  const float* obs = (const float*)d_in[0];   // [B, T, S] f32
  const float* lt  = (const float*)d_in[1];   // [S, S] f32
  float* out = (float*)d_out;                 // [B] f32
  crf_forward_wmma<<<dim3(CRF_B / 16), dim3(256), 0, stream>>>(obs, lt, out);
}
